// GDPModel_26749056319490
// MI455X (gfx1250) — matrix operand order repaired and checked by reading, not compiled, not run on hardware
//
#include <hip/hip_runtime.h>

// GAT 2-layer on MI455X (gfx1250, wave32).
// GEMMs via V_WMMA_F32_16X16X32_BF16 with bf16 hi/lo split (3 WMMAs per
// 16x16x32 step) for near-fp32 accuracy; edge softmax/scatter via atomics
// (the real roofline limit: ~GBs of gather/scatter at 23.3 TB/s HBM).

#define NN    50000
#define EE    800000
#define EP    850000   /* EE + NN self loops */
#define IN_F  128
#define HID1  32
#define H1    4
#define C1T   128      /* H1*HID1 */
#define OUTC  64
#define GG    64
#define EDF   16
#define SLOPE 0.2f

typedef __attribute__((ext_vector_type(16))) __bf16 v16bf;
typedef __attribute__((ext_vector_type(8)))  float  v8f;

union BF16x16 { v16bf v; unsigned short u[16]; };

__device__ __forceinline__ unsigned short f2bf(float f) {
  unsigned u = __float_as_uint(f);
  u += 0x7fffu + ((u >> 16) & 1u);           // round-to-nearest-even
  return (unsigned short)(u >> 16);
}
__device__ __forceinline__ float bf2f(unsigned short h) {
  return __uint_as_float(((unsigned)h) << 16);
}
// order-preserving float<->uint for atomicMax
__device__ __forceinline__ unsigned encf(float f) {
  unsigned u = __float_as_uint(f);
  return (u & 0x80000000u) ? ~u : (u | 0x80000000u);
}
__device__ __forceinline__ float decf(unsigned u) {
  return (u & 0x80000000u) ? __uint_as_float(u & 0x7fffffffu)
                           : __uint_as_float(~u);
}

// ---------------------------------------------------------------- zero
__global__ void k_zero(float* __restrict__ p, long n) {
  long i = (long)blockIdx.x * blockDim.x + threadIdx.x;
  if (i < n) p[i] = 0.f;
}

// ------------------------------------------------- edge_attr column sums
__global__ void k_colsum(const float* __restrict__ ea, float* __restrict__ colsum) {
  float loc[EDF];
#pragma unroll
  for (int k = 0; k < EDF; ++k) loc[k] = 0.f;
  long stride = (long)gridDim.x * blockDim.x;
  for (long r = (long)blockIdx.x * blockDim.x + threadIdx.x; r < EE; r += stride) {
    const float* row = ea + r * EDF;
#pragma unroll
    for (int k = 0; k < EDF; ++k) loc[k] += row[k];
  }
#pragma unroll
  for (int k = 0; k < EDF; ++k) atomicAdd(&colsum[k], loc[k]);
}

// -------- precompute w_e_att = We . att_edge^T  and self-loop fill scores
__global__ void k_prep(const float* __restrict__ We1, const float* __restrict__ ae1,
                       const float* __restrict__ We2, const float* __restrict__ ae2,
                       const float* __restrict__ colsum,
                       float* __restrict__ weatt1, float* __restrict__ fill1,
                       float* __restrict__ weatt2, float* __restrict__ fill2) {
  int tid = threadIdx.x;                       // 64 threads
  if (tid < 64) {
    int k = tid & 15, h = tid >> 4;            // weatt1[k][h]
    float a = 0.f;
#pragma unroll
    for (int c = 0; c < HID1; ++c) a += We1[k * C1T + h * HID1 + c] * ae1[h * HID1 + c];
    weatt1[k * H1 + h] = a;
  }
  if (tid < 16) {
    float a = 0.f;
#pragma unroll
    for (int c = 0; c < OUTC; ++c) a += We2[tid * OUTC + c] * ae2[c];
    weatt2[tid] = a;
  }
  __threadfence_block();
  __syncthreads();
  if (tid < H1) {
    float f = 0.f;
#pragma unroll
    for (int k = 0; k < EDF; ++k) f += (colsum[k] / (float)EE) * weatt1[k * H1 + tid];
    fill1[tid] = f;
  }
  if (tid == 4) {
    float f = 0.f;
#pragma unroll
    for (int k = 0; k < EDF; ++k) f += (colsum[k] / (float)EE) * weatt2[k];
    fill2[0] = f;
  }
}

// ------------------------------------------- WMMA GEMM, bf16 hi/lo split
// C[M x Nn] = A[M x K] * B[K x Nn], fp32 in/out. One wave per 16x16 tile.
__global__ __launch_bounds__(256)
void k_gemm_bf16x3(const float* __restrict__ A, const float* __restrict__ B,
                   float* __restrict__ C, int M, int K, int Nn) {
  const int wave = threadIdx.x >> 5;
  const int lane = threadIdx.x & 31;
  const int tilesM = M >> 4, tilesN = Nn >> 4;
  const int tile = blockIdx.x * 8 + wave;
  if (tile >= tilesM * tilesN) return;          // wave-uniform: EXEC stays full
  const int tm = (tile % tilesM) << 4;
  const int tn = (tile / tilesM) << 4;
  const int l15  = lane & 15;
  const int half = lane >> 4;                   // 0: lanes 0-15, 1: lanes 16-31
  const int mrow = tm + l15;
  const int ncol = tn + l15;
  v8f acc = {0.f, 0.f, 0.f, 0.f, 0.f, 0.f, 0.f, 0.f};

  for (int kk = 0; kk < K; kk += 32) {
    BF16x16 ahi, alo, bhi, blo;
    // A 16x32 bf16 layout: lanes<16 -> K {0..15}; lanes>=16 -> K {16..31}
    const float* arow = A + (size_t)mrow * K + kk + half * 16;
#pragma unroll
    for (int i = 0; i < 16; ++i) {
      int kof = i;
      float f = arow[kof];
      unsigned short hb = f2bf(f);
      ahi.u[i] = hb;
      alo.u[i] = f2bf(f - bf2f(hb));
    }
    // B 32x16 bf16 layout: lanes<16 -> K {0..15}; lanes>=16 -> K {16..31}, N=lane&15
    const float* bcol = B + (size_t)(kk + half * 16) * Nn + ncol;
#pragma unroll
    for (int i = 0; i < 16; ++i) {
      float f = bcol[(size_t)i * Nn];
      unsigned short hb = f2bf(f);
      bhi.u[i] = hb;
      blo.u[i] = f2bf(f - bf2f(hb));
    }
    acc = __builtin_amdgcn_wmma_f32_16x16x32_bf16(false, ahi.v, false, bhi.v,
                                                  (short)0, acc, false, false);
    acc = __builtin_amdgcn_wmma_f32_16x16x32_bf16(false, ahi.v, false, blo.v,
                                                  (short)0, acc, false, false);
    acc = __builtin_amdgcn_wmma_f32_16x16x32_bf16(false, alo.v, false, bhi.v,
                                                  (short)0, acc, false, false);
  }
  // D layout: element r -> row tm + r + 8*half, col = ncol
#pragma unroll
  for (int r = 0; r < 8; ++r)
    C[(size_t)(tm + r + half * 8) * Nn + ncol] = acc[r];
}

// ------------------------------- per-node attention scores a_src / a_dst
__global__ void k_node_scores(const float* __restrict__ hbuf,
                              const float* __restrict__ att_s,
                              const float* __restrict__ att_d,
                              float* __restrict__ asrc, float* __restrict__ adst,
                              int nNodes, int H, int C) {
  int i = blockIdx.x * blockDim.x + threadIdx.x;
  if (i >= nNodes * H) return;
  int n = i / H, h = i % H;
  const float* row = hbuf + (size_t)n * H * C + (size_t)h * C;
  float s = 0.f, d = 0.f;
  for (int c = 0; c < C; ++c) { s += row[c] * att_s[h * C + c]; d += row[c] * att_d[h * C + c]; }
  asrc[i] = s; adst[i] = d;
}

// ----------------- pass A: alpha = leaky_relu(...), atomic max per (dst,h)
__global__ void k_edge_alpha(const int* __restrict__ src, const int* __restrict__ dst,
                             const float* __restrict__ ea,
                             const float* __restrict__ weatt, const float* __restrict__ fill,
                             const float* __restrict__ asrc, const float* __restrict__ adst,
                             float* __restrict__ albuf, unsigned* __restrict__ amax, int H) {
  int e = blockIdx.x * blockDim.x + threadIdx.x;
  if (e >= EP) return;
  int s, d;
  float aeh[H1];
  if (e < EE) {
    s = src[e]; d = dst[e];
    const float* row = ea + (size_t)e * EDF;
    for (int h = 0; h < H; ++h) {
      float a = 0.f;
#pragma unroll
      for (int k = 0; k < EDF; ++k) a += row[k] * weatt[k * H + h];
      aeh[h] = a;
    }
  } else {
    s = d = e - EE;
    for (int h = 0; h < H; ++h) aeh[h] = fill[h];
  }
  for (int h = 0; h < H; ++h) {
    float a = asrc[(size_t)s * H + h] + adst[(size_t)d * H + h] + aeh[h];
    a = (a >= 0.f) ? a : SLOPE * a;
    albuf[(size_t)e * H + h] = a;
    atomicMax(&amax[(size_t)d * H + h], encf(a));
  }
}

// -------------------- pass B: ex = exp(alpha - max), atomic sum per (dst,h)
__global__ void k_edge_exp(const int* __restrict__ dst, float* __restrict__ albuf,
                           const unsigned* __restrict__ amax, float* __restrict__ den, int H) {
  int e = blockIdx.x * blockDim.x + threadIdx.x;
  if (e >= EP) return;
  int d = (e < EE) ? dst[e] : (e - EE);
  for (int h = 0; h < H; ++h) {
    float a  = albuf[(size_t)e * H + h];
    float ex = __expf(a - decf(amax[(size_t)d * H + h]));
    albuf[(size_t)e * H + h] = ex;
    atomicAdd(&den[(size_t)d * H + h], ex);
  }
}

// ------------- pass C: out[dst] += h[src] * coef  (one wave per edge)
__global__ __launch_bounds__(256)
void k_edge_msg(const int* __restrict__ src, const int* __restrict__ dst,
                const float* __restrict__ albuf, const float* __restrict__ den,
                const float* __restrict__ hbuf, float* __restrict__ outacc,
                int H, int Ctot, int Cc) {
  int wave = threadIdx.x >> 5, lane = threadIdx.x & 31;
  long e = (long)blockIdx.x * 8 + wave;
  if (e >= EP) return;                          // uniform per wave
  int s = (e < EE) ? src[e] : (int)(e - EE);
  int d = (e < EE) ? dst[e] : (int)(e - EE);
  for (int it = 0; it < Ctot / 32; ++it) {
    int c = it * 32 + lane;
    int h = c / Cc;
    float coef = albuf[(size_t)e * H + h] / (den[(size_t)d * H + h] + 1e-16f);
    atomicAdd(&outacc[(size_t)d * Ctot + c], hbuf[(size_t)s * Ctot + c] * coef);
  }
}

// --------------------------------------- act1 = elu(out1 + b1), in place
__global__ void k_bias_act(float* __restrict__ a, const float* __restrict__ b,
                           long n, int stride) {
  long i = (long)blockIdx.x * blockDim.x + threadIdx.x;
  if (i >= n) return;
  float v = a[i] + b[i % stride];
  a[i] = (v > 0.f) ? v : (__expf(v) - 1.f);
}

// ------------------------------- out2 + b2, scatter into per-graph pools
__global__ void k_bias_pool(const float* __restrict__ out2, const float* __restrict__ b2,
                            const int* __restrict__ batch,
                            float* __restrict__ pool, float* __restrict__ cnt) {
  long i = (long)blockIdx.x * blockDim.x + threadIdx.x;
  if (i >= (long)NN * OUTC) return;
  int n = (int)(i / OUTC), c = (int)(i % OUTC);
  int g = batch[n];
  atomicAdd(&pool[(size_t)g * OUTC + c], out2[i] + b2[c]);
  if (c == 0) atomicAdd(&cnt[g], 1.f);
}

__global__ void k_final(const float* __restrict__ pool, const float* __restrict__ cnt,
                        float* __restrict__ out) {
  int i = blockIdx.x * blockDim.x + threadIdx.x;
  if (i >= GG * OUTC) return;
  out[i] = pool[i] / fmaxf(cnt[i / OUTC], 1.f);
}

// ====================================================================
extern "C" void kernel_launch(void* const* d_in, const int* in_sizes, int n_in,
                              void* d_out, int out_size, void* d_ws, size_t ws_size,
                              hipStream_t stream) {
  (void)in_sizes; (void)n_in; (void)out_size; (void)ws_size;
  const float* x        = (const float*)d_in[0];
  const int*   ei       = (const int*)d_in[1];
  const float* ea       = (const float*)d_in[2];
  const int*   batch    = (const int*)d_in[3];
  const float* W1       = (const float*)d_in[4];
  const float* att_src1 = (const float*)d_in[5];
  const float* att_dst1 = (const float*)d_in[6];
  const float* We1      = (const float*)d_in[7];
  const float* att_e1   = (const float*)d_in[8];
  const float* b1       = (const float*)d_in[9];
  const float* W2       = (const float*)d_in[10];
  const float* att_src2 = (const float*)d_in[11];
  const float* att_dst2 = (const float*)d_in[12];
  const float* We2      = (const float*)d_in[13];
  const float* att_e2   = (const float*)d_in[14];
  const float* b2       = (const float*)d_in[15];
  const int* src = ei;
  const int* dst = ei + EE;
  float* out = (float*)d_out;

  char* base = (char*)d_ws;
  size_t off = 0;
  auto alloc = [&](size_t bytes) -> void* {
    void* p = base + off;
    off = (off + bytes + 255) & ~(size_t)255;
    return p;
  };
  // --- zero-initialized region (contiguous; cleared every launch) ---
  float*    act1   = (float*)alloc((size_t)NN * C1T * 4);   // out1 accum -> act1
  float*    out2   = (float*)alloc((size_t)NN * OUTC * 4);
  float*    den1   = (float*)alloc((size_t)NN * H1 * 4);
  float*    den2   = (float*)alloc((size_t)NN * 4);
  unsigned* amax1  = (unsigned*)alloc((size_t)NN * H1 * 4);
  unsigned* amax2  = (unsigned*)alloc((size_t)NN * 4);
  float*    colsum = (float*)alloc(EDF * 4);
  float*    pool   = (float*)alloc((size_t)GG * OUTC * 4);
  float*    cnt    = (float*)alloc((size_t)GG * 4);
  long zero_elems  = (long)(off / 4);
  // --- non-zeroed scratch ---
  float* h1     = (float*)alloc((size_t)NN * C1T * 4);
  float* h2     = (float*)alloc((size_t)NN * OUTC * 4);
  float* asrc1  = (float*)alloc((size_t)NN * H1 * 4);
  float* adst1  = (float*)alloc((size_t)NN * H1 * 4);
  float* asrc2  = (float*)alloc((size_t)NN * 4);
  float* adst2  = (float*)alloc((size_t)NN * 4);
  float* ex1    = (float*)alloc((size_t)EP * H1 * 4);
  float* ex2    = (float*)alloc((size_t)EP * 4);
  float* weatt1 = (float*)alloc(EDF * H1 * 4);
  float* fill1  = (float*)alloc(16);
  float* weatt2 = (float*)alloc(EDF * 4);
  float* fill2  = (float*)alloc(16);

  k_zero<<<(unsigned)((zero_elems + 255) / 256), 256, 0, stream>>>((float*)d_ws, zero_elems);
  k_colsum<<<16, 256, 0, stream>>>(ea, colsum);
  k_prep<<<1, 64, 0, stream>>>(We1, att_e1, We2, att_e2, colsum, weatt1, fill1, weatt2, fill2);

  // -------- layer 1 --------
  { int tiles = (NN / 16) * (C1T / 16);
    k_gemm_bf16x3<<<(tiles + 7) / 8, 256, 0, stream>>>(x, W1, h1, NN, IN_F, C1T); }
  k_node_scores<<<(NN * H1 + 255) / 256, 256, 0, stream>>>(h1, att_src1, att_dst1,
                                                           asrc1, adst1, NN, H1, HID1);
  k_edge_alpha<<<(EP + 255) / 256, 256, 0, stream>>>(src, dst, ea, weatt1, fill1,
                                                     asrc1, adst1, ex1, amax1, H1);
  k_edge_exp<<<(EP + 255) / 256, 256, 0, stream>>>(dst, ex1, amax1, den1, H1);
  k_edge_msg<<<(EP + 7) / 8, 256, 0, stream>>>(src, dst, ex1, den1, h1, act1, H1, C1T, HID1);
  k_bias_act<<<(unsigned)(((long)NN * C1T + 255) / 256), 256, 0, stream>>>(act1, b1,
                                                             (long)NN * C1T, C1T);
  // -------- layer 2 --------
  { int tiles = (NN / 16) * (OUTC / 16);
    k_gemm_bf16x3<<<(tiles + 7) / 8, 256, 0, stream>>>(act1, W2, h2, NN, C1T, OUTC); }
  k_node_scores<<<(NN + 255) / 256, 256, 0, stream>>>(h2, att_src2, att_dst2,
                                                      asrc2, adst2, NN, 1, OUTC);
  k_edge_alpha<<<(EP + 255) / 256, 256, 0, stream>>>(src, dst, ea, weatt2, fill2,
                                                     asrc2, adst2, ex2, amax2, 1);
  k_edge_exp<<<(EP + 255) / 256, 256, 0, stream>>>(dst, ex2, amax2, den2, 1);
  k_edge_msg<<<(EP + 7) / 8, 256, 0, stream>>>(src, dst, ex2, den2, h2, out2, 1, OUTC, OUTC);
  // -------- pooling --------
  k_bias_pool<<<(unsigned)(((long)NN * OUTC + 255) / 256), 256, 0, stream>>>(out2, b2,
                                                                             batch, pool, cnt);
  k_final<<<(GG * OUTC + 255) / 256, 256, 0, stream>>>(pool, cnt, out);
}
